// Extractor_36721970381000
// MI455X (gfx1250) — matrix operand order repaired and checked by reading, not compiled, not run on hardware
//
#include <hip/hip_runtime.h>

typedef __attribute__((ext_vector_type(16))) __bf16 v16bf;
typedef __attribute__((ext_vector_type(8)))  float  v8f;

#define BATCH   16384
#define NCODES  1024
#define KD      2048          // Q_LEN * D_MODEL
#define DECAYF  0.99f
#define OMDF    (1.0f - DECAYF)
#define EPSF    1e-5f

// ---------------- init: out_edw = decay*ema_dw ; counts = 0 ----------------
__global__ void vq_init(const float* __restrict__ ema_dw,
                        float* __restrict__ out_edw,
                        float* __restrict__ counts) {
  int i = blockIdx.x * blockDim.x + threadIdx.x;     // 0 .. NCODES*KD-1
  out_edw[i] = DECAYF * ema_dw[i];
  if (i < NCODES) counts[i] = 0.0f;
}

// ---------------- per-code squared norms ----------------
__global__ void vq_gnorm(const float* __restrict__ ctx, float* __restrict__ gnorm) {
  __shared__ float red[256];
  const int n = blockIdx.x;
  const float* row = ctx + (long)n * KD;
  float s = 0.f;
  for (int j = threadIdx.x; j < KD; j += 256) { float v = row[j]; s = fmaf(v, v, s); }
  red[threadIdx.x] = s; __syncthreads();
  for (int st = 128; st > 0; st >>= 1) {
    if (threadIdx.x < st) red[threadIdx.x] += red[threadIdx.x + st];
    __syncthreads();
  }
  if (threadIdx.x == 0) gnorm[n] = red[0];
}

// ---------------- WMMA distance GEMM + fused argmin ----------------
// 128 rows / workgroup (8 waves x 16-row strips). Per wave: 4 live 16x16 C
// tiles (64 codes), K=2048 in steps of 32 with bf16 hi/lo split:
//   dot_f32 ~= a_hi*b_hi + a_hi*b_lo + a_lo*b_hi   (3 WMMAs / tile / k-step)
// Context k-slice staged once per WG in LDS as pre-split bf16 (shared by 8 waves).
__launch_bounds__(256)
__global__ void vq_argmin(const float* __restrict__ q,
                          const float* __restrict__ ctx,
                          const float* __restrict__ gnorm,
                          int* __restrict__ idx_out,
                          float* __restrict__ qld_out) {
  __shared__ __attribute__((aligned(32))) __bf16 lbhi[64 * 32];
  __shared__ __attribute__((aligned(32))) __bf16 lblo[64 * 32];

  const int tid     = threadIdx.x;
  const int wave    = tid >> 5;
  const int lane    = tid & 31;
  const int laneLow = lane & 15;
  const int khalf   = lane >> 4;

  const int rowBase = blockIdx.x * 128 + wave * 16;
  const float* qrow = q + (long)(rowBase + laneLow) * KD;

  float best[8]; int bidx[8];
#pragma unroll
  for (int r = 0; r < 8; ++r) { best[r] = __builtin_inff(); bidx[r] = 0; }
  float qn = 0.f;                                   // per-row ||q||^2 (partial)

  for (int nc = 0; nc < 16; ++nc) {                 // 16 chunks of 64 codes
    const int n0 = nc * 64;
    v8f acc[4];
#pragma unroll
    for (int t = 0; t < 4; ++t)
#pragma unroll
      for (int r = 0; r < 8; ++r) acc[t][r] = 0.0f;

    for (int ks = 0; ks < 64; ++ks) {               // K loop, 32 per step
      const int k0 = ks * 32;
      __syncthreads();
      { // cooperative stage of B slice [64 codes][32 k] -> split bf16 hi/lo
        const int col = tid >> 2;                   // 0..63
        const int kk  = (tid & 3) * 8;              // 0,8,16,24
        const float4* src = (const float4*)(ctx + (long)(n0 + col) * KD + k0 + kk);
        float4 f0 = src[0], f1 = src[1];
        float vals[8] = {f0.x, f0.y, f0.z, f0.w, f1.x, f1.y, f1.z, f1.w};
        __bf16* hh = &lbhi[col * 32 + kk];
        __bf16* ll = &lblo[col * 32 + kk];
#pragma unroll
        for (int j = 0; j < 8; ++j) {
          __bf16 h = (__bf16)vals[j];
          hh[j] = h;
          ll[j] = (__bf16)(vals[j] - (float)h);
        }
      }
      __syncthreads();

      // A fragment (16x32): lane<16 holds K {0..15}, lane>=16 {16..31}
      const float4* p0 = (const float4*)(qrow + k0 + khalf * 16);
      float4 a0 = p0[0], a1 = p0[1], a2 = p0[2], a3 = p0[3];
      float av[16] = {a0.x, a0.y, a0.z, a0.w, a1.x, a1.y, a1.z, a1.w,
                      a2.x, a2.y, a2.z, a2.w, a3.x, a3.y, a3.z, a3.w};
      v16bf ahi, alo;
#pragma unroll
      for (int j = 0; j < 16; ++j) {
        __bf16 h = (__bf16)av[j];
        ahi[j] = h;
        alo[j] = (__bf16)(av[j] - (float)h);
      }
      if (nc == 0) {                                // fuse ||q||^2 on first pass
#pragma unroll
        for (int j = 0; j < 16; ++j) qn = fmaf(av[j], av[j], qn);
      }

#pragma unroll
      for (int t = 0; t < 4; ++t) {                 // 4 code tiles share A
        const int colLocal = t * 16 + laneLow;
        v16bf bh = *(const v16bf*)&lbhi[colLocal * 32 + khalf * 16];
        v16bf bl = *(const v16bf*)&lblo[colLocal * 32 + khalf * 16];
        acc[t] = __builtin_amdgcn_wmma_f32_16x16x32_bf16(false, alo, false, bh, (short)0, acc[t], false, false);
        acc[t] = __builtin_amdgcn_wmma_f32_16x16x32_bf16(false, ahi, false, bl, (short)0, acc[t], false, false);
        acc[t] = __builtin_amdgcn_wmma_f32_16x16x32_bf16(false, ahi, false, bh, (short)0, acc[t], false, false);
      }
    }

    // epilogue: candidates = ||g||^2 - 2*dot (||q||^2 constant per row)
#pragma unroll
    for (int t = 0; t < 4; ++t) {
      const int n = n0 + t * 16 + laneLow;
      const float gn = gnorm[n];
#pragma unroll
      for (int r = 0; r < 8; ++r) {
        float v = fmaf(-2.0f, acc[t][r], gn);
        if (v < best[r]) { best[r] = v; bidx[r] = n; }  // n increasing: first-min kept
      }
    }
  }

  qn += __shfl_xor(qn, 16, 32);                     // complete per-row ||q||^2

#pragma unroll
  for (int r = 0; r < 8; ++r) {
    float v = best[r]; int i = bidx[r];
#pragma unroll
    for (int off = 8; off >= 1; off >>= 1) {        // butterfly within 16-lane group
      float ov = __shfl_xor(v, off, 32);
      int   oi = __shfl_xor(i, off, 32);
      if (ov < v || (ov == v && oi < i)) { v = ov; i = oi; }
    }
    const int m = r + 8 * khalf;                    // C-tile row this group reduced
    if (laneLow == m) {                             // writer lane holds row m's qn
      const int row = rowBase + m;
      idx_out[row] = i;
      qld_out[row] = fmaxf(qn + v, 0.0f) * (1.0f / (float)KD);
    }
  }
}

// ---------------- scatter: edw += (1-decay)*q[b] into code row; counts ------
__global__ void vq_scatter(const float* __restrict__ q,
                           const int* __restrict__ idx,
                           float* __restrict__ out_edw,
                           float* __restrict__ counts) {
  const int b = blockIdx.x;
  const int code = idx[b];
  const float* src = q + (long)b * KD;
  float* dst = out_edw + (long)code * KD;
  for (int j = threadIdx.x; j < KD; j += 256)
    atomicAdd(&dst[j], OMDF * src[j]);
  if (threadIdx.x == 0) atomicAdd(&counts[code], 1.0f);
}

// ---------------- ec: EMA counts + Laplace smoothing (single block) ---------
__global__ void vq_ec(const float* __restrict__ ema_count,
                      const float* __restrict__ counts,
                      float* __restrict__ out_ec,
                      float* __restrict__ ws_ec) {
  __shared__ float red[NCODES];
  const int i = threadIdx.x;
  float ecr = DECAYF * ema_count[i] + OMDF * counts[i];
  red[i] = ecr; __syncthreads();
  for (int s = 512; s > 0; s >>= 1) {
    if (i < s) red[i] += red[i + s];
    __syncthreads();
  }
  const float n = red[0];
  const float ec = (ecr + EPSF) / (n + (float)KD * EPSF) * n;
  out_ec[i] = ec;
  ws_ec[i]  = ec;
}

// ---------------- final: context_new = edw/ec ; out_context == context_new --
__global__ void vq_final(const float* __restrict__ out_edw,
                         const float* __restrict__ ws_ec,
                         float* __restrict__ out_cn,
                         float* __restrict__ out_ctx) {
  const int i = blockIdx.x * blockDim.x + threadIdx.x;
  const int code = i >> 11;                          // / KD
  const float c = out_edw[i] / ws_ec[code];
  out_cn[i]  = c;
  out_ctx[i] = c;
}

extern "C" void kernel_launch(void* const* d_in, const int* in_sizes, int n_in,
                              void* d_out, int out_size, void* d_ws, size_t ws_size,
                              hipStream_t stream) {
  (void)in_sizes; (void)n_in; (void)out_size; (void)ws_size;
  const float* q       = (const float*)d_in[0];
  const float* ctx     = (const float*)d_in[1];
  const float* ema_cnt = (const float*)d_in[2];
  const float* ema_dw  = (const float*)d_in[3];

  // outputs concatenated in return order: qld, out_context, ec, edw, context_new
  float* out_qld = (float*)d_out;
  float* out_ctx = out_qld + BATCH;
  float* out_ec  = out_ctx + (long)NCODES * KD;
  float* out_edw = out_ec + NCODES;
  float* out_cn  = out_edw + (long)NCODES * KD;

  // workspace: indices, counts, gnorm, ec  (~76 KB)
  int*   ws_idx    = (int*)d_ws;
  float* ws_counts = (float*)d_ws + BATCH;
  float* ws_gnorm  = ws_counts + NCODES;
  float* ws_ec     = ws_gnorm + NCODES;

  vq_init   <<<(NCODES * KD) / 256, 256, 0, stream>>>(ema_dw, out_edw, ws_counts);
  vq_gnorm  <<<NCODES,              256, 0, stream>>>(ctx, ws_gnorm);
  vq_argmin <<<BATCH / 128,         256, 0, stream>>>(q, ctx, ws_gnorm, ws_idx, out_qld);
  vq_scatter<<<BATCH,               256, 0, stream>>>(q, ws_idx, out_edw, ws_counts);
  vq_ec     <<<1,                  1024, 0, stream>>>(ema_cnt, ws_counts, out_ec, ws_ec);
  vq_final  <<<(NCODES * KD) / 256, 256, 0, stream>>>(out_edw, ws_ec, out_cn, out_ctx);
}
